// DCGRUCell_35837207118644
// MI455X (gfx1250) — compile-verified, matrix operand order checked by reading, not run on hardware
//
#include <hip/hip_runtime.h>
#include <hip/hip_bf16.h>
#include <math.h>

// ---------------------------------------------------------------------------
// DCGRU cell for MI455X (gfx1250): bf16 WMMA 16x16x32, f32 accumulate.
// Dims: B=32, N=2048, D_in=64, D_h=128, in_total=192, K=3, FT=7*192=1344.
// ---------------------------------------------------------------------------

typedef __attribute__((ext_vector_type(16))) __bf16 v16bf;
typedef __attribute__((ext_vector_type(8)))  __bf16 v8bf;
typedef __attribute__((ext_vector_type(8)))  float  v8f;

#define NN   2048
#define BB   32
#define DIN  64
#define DH   128
#define SL   192      // in_total
#define FT   1344     // (2K+1)*in_total
#define MROWS (BB*NN) // 65536

// Load one 16-bf16 WMMA A/B fragment for lane (r,half) from a K-contiguous row.
// ISA layout: element e <-> k = kBase + (e<8 ? e : e+8) + 8*half  => two b128 loads.
__device__ __forceinline__ v16bf load_frag(const __bf16* rowPtr, int kBase, int half) {
    const v8bf lo = *reinterpret_cast<const v8bf*>(rowPtr + kBase + 8 * half);
    const v8bf hi = *reinterpret_cast<const v8bf*>(rowPtr + kBase + 16 + 8 * half);
    v16bf f;
#pragma unroll
    for (int e = 0; e < 8; ++e) { f[e] = lo[e]; f[e + 8] = hi[e]; }
    return f;
}

// ---------------- f32 -> bf16 conversion --------------------------------------
__global__ void __launch_bounds__(256) cvt_bf16_kernel(const float* __restrict__ src,
                                                       __bf16* __restrict__ dst, int n) {
    int i = blockIdx.x * blockDim.x + threadIdx.x;
    if (i < n) dst[i] = (__bf16)src[i];
}

// ---------------- pack x_h = [x, h] into featT slice0 + feat slice0 -----------
__global__ void __launch_bounds__(256) pack_xh_kernel(const float* __restrict__ x,
                                                      const float* __restrict__ h,
                                                      __bf16* __restrict__ feat,
                                                      __bf16* __restrict__ featT) {
    int idx = blockIdx.x * blockDim.x + threadIdx.x;          // B*N*SL threads
    int f = idx % SL;
    int n = (idx / SL) % NN;
    int b = idx / (SL * NN);
    if (b >= BB) return;
    float v = (f < DIN) ? x[((size_t)b * NN + n) * DIN + f]
                        : h[((size_t)b * NN + n) * DH + (f - DIN)];
    __bf16 bv = (__bf16)v;
    feat[((size_t)b * NN + n) * FT + f] = bv;                 // node-major slice 0
    featT[((size_t)b * FT + f) * NN + n] = bv;                // feature-major slice 0
}

// ---------------- pack x_rh = [x, r*h] into featT slice0 + feat slice0 --------
__global__ void __launch_bounds__(256) pack_xrh_kernel(const float* __restrict__ x,
                                                       const float* __restrict__ rg,
                                                       const float* __restrict__ h,
                                                       __bf16* __restrict__ feat,
                                                       __bf16* __restrict__ featT) {
    int idx = blockIdx.x * blockDim.x + threadIdx.x;
    int f = idx % SL;
    int n = (idx / SL) % NN;
    int b = idx / (SL * NN);
    if (b >= BB) return;
    float v;
    if (f < DIN) v = x[((size_t)b * NN + n) * DIN + f];
    else {
        size_t o = ((size_t)b * NN + n) * DH + (f - DIN);
        v = rg[o] * h[o];
    }
    __bf16 bv = (__bf16)v;
    feat[((size_t)b * NN + n) * FT + f] = bv;
    featT[((size_t)b * FT + f) * NN + n] = bv;
}

// ---------------- one diffusion hop: Y^T = X^T * W^T (per batch) --------------
// A = featT slice sliceIn  [192 x 2048]  (rows K-contiguous)
// B cols = rows of W       [2048 x 2048] (K-contiguous)
// Wave computes a 16(f) x 64(node) strip = 4 C tiles, k-loop 2048/32.
// C stored transposed into featT slice sliceOut and node-major into feat.
__global__ void __launch_bounds__(256) diff_hop_kernel(const __bf16* __restrict__ W,
                                                       __bf16* featT,
                                                       __bf16* __restrict__ feat,
                                                       int sliceIn, int sliceOut) {
    int tid  = blockIdx.x * blockDim.x + threadIdx.x;
    int wid  = tid >> 5;                  // 12288 waves: 32 batches * 12 fTiles * 32 strips
    int lane = threadIdx.x & 31;
    int b      = wid / 384;
    int t      = wid - b * 384;
    int fTile  = t % 12;
    int iStrip = t / 12;
    int fBase  = fTile * 16;
    int iBase  = iStrip * 64;
    int half   = lane >> 4;
    int r      = lane & 15;

    const __bf16* Arow  = featT + ((size_t)b * FT + (size_t)sliceIn * SL + fBase + r) * NN;
    const __bf16* Brow0 = W + (size_t)(iBase + r) * NN;

    v8f c0 = {}, c1 = {}, c2 = {}, c3 = {};
    for (int kb = 0; kb < NN; kb += 32) {
        v16bf a  = load_frag(Arow, kb, half);
        v16bf b0 = load_frag(Brow0,                     kb, half);
        v16bf b1 = load_frag(Brow0 + (size_t)16 * NN,   kb, half);
        v16bf b2 = load_frag(Brow0 + (size_t)32 * NN,   kb, half);
        v16bf b3 = load_frag(Brow0 + (size_t)48 * NN,   kb, half);
        c0 = __builtin_amdgcn_wmma_f32_16x16x32_bf16(false, a, false, b0, (short)0, c0, false, false);
        c1 = __builtin_amdgcn_wmma_f32_16x16x32_bf16(false, a, false, b1, (short)0, c1, false, false);
        c2 = __builtin_amdgcn_wmma_f32_16x16x32_bf16(false, a, false, b2, (short)0, c2, false, false);
        c3 = __builtin_amdgcn_wmma_f32_16x16x32_bf16(false, a, false, b3, (short)0, c3, false, false);
    }

    __bf16* outT = featT + ((size_t)b * FT + (size_t)sliceOut * SL) * NN;
    __bf16* outN = feat + (size_t)b * NN * FT;
    v8f cs[4] = {c0, c1, c2, c3};
#pragma unroll
    for (int j = 0; j < 4; ++j) {
        int n = iBase + j * 16 + r;
        v8bf pk;
#pragma unroll
        for (int v = 0; v < 8; ++v) {
            int f = fBase + v + 8 * half;                  // C: m = v + 8*half
            __bf16 bv = (__bf16)cs[j][v];
            outT[(size_t)f * NN + n] = bv;                 // feature-major
            pk[v] = bv;
        }
        // node-major: 8 consecutive f values -> one b128 store
        *reinterpret_cast<v8bf*>(outN + (size_t)n * FT + sliceOut * SL + fBase + 8 * half) = pk;
    }
}

// ---------------- gate MLP: sigmoid(feat @ Wg^T + b) --------------------------
// Wave: 16 rows x 64 cols, k-loop 1344/32 = 42. 8192 waves.
__global__ void __launch_bounds__(256) mlp_gate_kernel(const __bf16* __restrict__ feat,
                                                       const __bf16* __restrict__ Wg,
                                                       const float* __restrict__ bias,
                                                       float* __restrict__ out) {
    int tid  = blockIdx.x * blockDim.x + threadIdx.x;
    int wid  = tid >> 5;
    int lane = threadIdx.x & 31;
    int rowBase = (wid >> 1) * 16;
    int colBase = (wid & 1) * 64;
    int half = lane >> 4, r = lane & 15;

    const __bf16* Arow  = feat + (size_t)(rowBase + r) * FT;
    const __bf16* Brow0 = Wg + (size_t)(colBase + r) * FT;

    v8f c0 = {}, c1 = {}, c2 = {}, c3 = {};
    for (int kb = 0; kb < FT; kb += 32) {
        v16bf a  = load_frag(Arow, kb, half);
        v16bf b0 = load_frag(Brow0,                     kb, half);
        v16bf b1 = load_frag(Brow0 + (size_t)16 * FT,   kb, half);
        v16bf b2 = load_frag(Brow0 + (size_t)32 * FT,   kb, half);
        v16bf b3 = load_frag(Brow0 + (size_t)48 * FT,   kb, half);
        c0 = __builtin_amdgcn_wmma_f32_16x16x32_bf16(false, a, false, b0, (short)0, c0, false, false);
        c1 = __builtin_amdgcn_wmma_f32_16x16x32_bf16(false, a, false, b1, (short)0, c1, false, false);
        c2 = __builtin_amdgcn_wmma_f32_16x16x32_bf16(false, a, false, b2, (short)0, c2, false, false);
        c3 = __builtin_amdgcn_wmma_f32_16x16x32_bf16(false, a, false, b3, (short)0, c3, false, false);
    }
    v8f cs[4] = {c0, c1, c2, c3};
#pragma unroll
    for (int j = 0; j < 4; ++j) {
        int n = colBase + j * 16 + r;
        float bn = bias[n];
#pragma unroll
        for (int v = 0; v < 8; ++v) {
            int m = v + 8 * half;
            float val = cs[j][v] + bn;
            out[(size_t)(rowBase + m) * DH + n] = 1.0f / (1.0f + __expf(-val));
        }
    }
}

// ---------------- final: out = (1-z)*h + z*tanh(feat @ Wn^T + bn) -------------
__global__ void __launch_bounds__(256) mlp_final_kernel(const __bf16* __restrict__ feat,
                                                        const __bf16* __restrict__ Wn,
                                                        const float* __restrict__ bias,
                                                        const float* __restrict__ z,
                                                        const float* __restrict__ h,
                                                        float* __restrict__ out) {
    int tid  = blockIdx.x * blockDim.x + threadIdx.x;
    int wid  = tid >> 5;
    int lane = threadIdx.x & 31;
    int rowBase = (wid >> 1) * 16;
    int colBase = (wid & 1) * 64;
    int half = lane >> 4, r = lane & 15;

    const __bf16* Arow  = feat + (size_t)(rowBase + r) * FT;
    const __bf16* Brow0 = Wn + (size_t)(colBase + r) * FT;

    v8f c0 = {}, c1 = {}, c2 = {}, c3 = {};
    for (int kb = 0; kb < FT; kb += 32) {
        v16bf a  = load_frag(Arow, kb, half);
        v16bf b0 = load_frag(Brow0,                     kb, half);
        v16bf b1 = load_frag(Brow0 + (size_t)16 * FT,   kb, half);
        v16bf b2 = load_frag(Brow0 + (size_t)32 * FT,   kb, half);
        v16bf b3 = load_frag(Brow0 + (size_t)48 * FT,   kb, half);
        c0 = __builtin_amdgcn_wmma_f32_16x16x32_bf16(false, a, false, b0, (short)0, c0, false, false);
        c1 = __builtin_amdgcn_wmma_f32_16x16x32_bf16(false, a, false, b1, (short)0, c1, false, false);
        c2 = __builtin_amdgcn_wmma_f32_16x16x32_bf16(false, a, false, b2, (short)0, c2, false, false);
        c3 = __builtin_amdgcn_wmma_f32_16x16x32_bf16(false, a, false, b3, (short)0, c3, false, false);
    }
    v8f cs[4] = {c0, c1, c2, c3};
#pragma unroll
    for (int j = 0; j < 4; ++j) {
        int n = colBase + j * 16 + r;
        float bn = bias[n];
#pragma unroll
        for (int v = 0; v < 8; ++v) {
            int m = v + 8 * half;
            size_t row = (size_t)(rowBase + m);
            float nt = tanhf(cs[j][v] + bn);
            float zv = z[row * DH + n];
            float hv = h[row * DH + n];
            out[row * DH + n] = (1.0f - zv) * hv + zv * nt;
        }
    }
}

// ---------------------------------------------------------------------------
extern "C" void kernel_launch(void* const* d_in, const int* in_sizes, int n_in,
                              void* d_out, int out_size, void* d_ws, size_t ws_size,
                              hipStream_t stream) {
    const float* x  = (const float*)d_in[0];
    const float* h  = (const float*)d_in[1];
    const float* Wf = (const float*)d_in[2];
    const float* Wb = (const float*)d_in[3];
    const float* Wr = (const float*)d_in[4];
    const float* br = (const float*)d_in[5];
    const float* Wz = (const float*)d_in[6];
    const float* bz = (const float*)d_in[7];
    const float* Wn = (const float*)d_in[8];
    const float* bn = (const float*)d_in[9];
    float* out = (float*)d_out;

    char* ws = (char*)d_ws;
    size_t off = 0;
    auto carve = [&](size_t bytes) -> char* {
        char* p = ws + off;
        off = (off + bytes + 255) & ~(size_t)255;
        return p;
    };
    __bf16* Wf_bf  = (__bf16*)carve((size_t)NN * NN * 2);
    __bf16* Wb_bf  = (__bf16*)carve((size_t)NN * NN * 2);
    __bf16* Wr_bf  = (__bf16*)carve((size_t)DH * FT * 2);
    __bf16* Wz_bf  = (__bf16*)carve((size_t)DH * FT * 2);
    __bf16* Wn_bf  = (__bf16*)carve((size_t)DH * FT * 2);
    __bf16* featT  = (__bf16*)carve((size_t)BB * FT * NN * 2);   // feature-major
    __bf16* feat   = (__bf16*)carve((size_t)BB * NN * FT * 2);   // node-major
    float*  r_buf  = (float*)carve((size_t)MROWS * DH * 4);
    float*  z_buf  = (float*)carve((size_t)MROWS * DH * 4);

    auto cvt = [&](const float* s, __bf16* d, int n) {
        cvt_bf16_kernel<<<(n + 255) / 256, 256, 0, stream>>>(s, d, n);
    };
    cvt(Wf, Wf_bf, NN * NN);
    cvt(Wb, Wb_bf, NN * NN);
    cvt(Wr, Wr_bf, DH * FT);
    cvt(Wz, Wz_bf, DH * FT);
    cvt(Wn, Wn_bf, DH * FT);

    const int packBlocks = (BB * NN * SL + 255) / 256;           // 49152
    const int hopBlocks  = (BB * 12 * 32 * 32) / 256;            // 1536 (12288 waves)
    const int mlpBlocks  = ((MROWS / 16) * 2 * 32) / 256;        // 1024 (8192 waves)

    auto hops = [&]() {
        diff_hop_kernel<<<hopBlocks, 256, 0, stream>>>(Wf_bf, featT, feat, 0, 1);
        diff_hop_kernel<<<hopBlocks, 256, 0, stream>>>(Wf_bf, featT, feat, 1, 2);
        diff_hop_kernel<<<hopBlocks, 256, 0, stream>>>(Wf_bf, featT, feat, 2, 3);
        diff_hop_kernel<<<hopBlocks, 256, 0, stream>>>(Wb_bf, featT, feat, 0, 4);
        diff_hop_kernel<<<hopBlocks, 256, 0, stream>>>(Wb_bf, featT, feat, 4, 5);
        diff_hop_kernel<<<hopBlocks, 256, 0, stream>>>(Wb_bf, featT, feat, 5, 6);
    };

    // Pass 1: feat_xh -> r, z
    pack_xh_kernel<<<packBlocks, 256, 0, stream>>>(x, h, feat, featT);
    hops();
    mlp_gate_kernel<<<mlpBlocks, 256, 0, stream>>>(feat, Wr_bf, br, r_buf);
    mlp_gate_kernel<<<mlpBlocks, 256, 0, stream>>>(feat, Wz_bf, bz, z_buf);

    // Pass 2: feat_rh -> n_tilde -> combine
    pack_xrh_kernel<<<packBlocks, 256, 0, stream>>>(x, r_buf, h, feat, featT);
    hops();
    mlp_final_kernel<<<mlpBlocks, 256, 0, stream>>>(feat, Wn_bf, bn, z_buf, h, out);
}
